// InterventionalAttention_2637109919795
// MI455X (gfx1250) — matrix operand order repaired and checked by reading, not compiled, not run on hardware
//
#include <hip/hip_runtime.h>

// ---------------------------------------------------------------------------
// InterventionalAttention for MI455X (gfx1250, wave32, WMMA bf16 16x16x32)
//   B=2, L=2048, D=1024, H=16, dk=64
//   y = softmax(QK^T/8) with prob column 0 halved (no renorm), then @V, @Wo^T
// GEMMs: 64x64 per-wave tiles; fully-unrolled epilogue (no acc spills).
// ---------------------------------------------------------------------------

typedef __bf16 bf16_t;
typedef bf16_t v8bf  __attribute__((ext_vector_type(8)));
typedef bf16_t v16bf __attribute__((ext_vector_type(16)));
typedef float  v8f   __attribute__((ext_vector_type(8)));

#define D_MODEL 1024
#define N_HEADS 16
#define DK      64
#define BATCH   2
#define SEQ     2048
#define ROWS    (BATCH * SEQ)   // 4096

static __device__ __forceinline__ v16bf cat16(v8bf lo, v8bf hi) {
  return __builtin_shufflevector(lo, hi, 0,1,2,3,4,5,6,7,8,9,10,11,12,13,14,15);
}

// A fragment (16x32 bf16, row-major source, row stride ld).
// ISA layout: lane m=l&15 holds K chunks [h*8, h*8+8) and [16+h*8, ...), h=l>>4.
static __device__ __forceinline__ v16bf load_a(const bf16_t* base, int ld, int lane) {
  const bf16_t* p = base + (size_t)(lane & 15) * ld + (lane >> 4) * 8;
  v8bf lo = *(const v8bf*)p;
  v8bf hi = *(const v8bf*)(p + 16);
  return cat16(lo, hi);
}

// B fragment (32x16 bf16) where column n is contiguous over k with stride ld
// between columns (source stored [n][k] row-major).
// ISA layout: lane n=l&15 holds K chunks [h*8, h*8+8) and [16+h*8, ...), h=l>>4.
static __device__ __forceinline__ v16bf load_b(const bf16_t* base, int ld, int lane) {
  const bf16_t* p = base + (size_t)(lane & 15) * ld + (lane >> 4) * 8;
  v8bf lo = *(const v8bf*)p;
  v8bf hi = *(const v8bf*)(p + 16);
  return cat16(lo, hi);
}

static __device__ __forceinline__ v8f wmma_bf16(v16bf a, v16bf b, v8f c) {
  return __builtin_amdgcn_wmma_f32_16x16x32_bf16(
      /*neg_a=*/false, a, /*neg_b=*/false, b,
      /*c_mod=*/(short)0, c, /*reuse_a=*/false, /*reuse_b=*/false);
}

// --------------------------- fp32 -> bf16 ----------------------------------
__global__ void f32_to_bf16_kernel(const float* __restrict__ in,
                                   bf16_t* __restrict__ out, int n) {
  int i = blockIdx.x * blockDim.x + threadIdx.x;
  if (i < n) out[i] = (bf16_t)in[i];
}

// --------------------------- GEMM: C = A @ W^T + bias ----------------------
// A: [M,K] bf16 row-major.  W: [N,K] bf16 row-major (W rows == B columns).
// 128 threads = 4 waves (2x2), 64x64 tile per wave, 128x128 per block.
// mode 0: float out, row-major [M,N]            (final projection)
// mode 1: bf16 out, [B,H,L,dk] head-split       (Q, K)
// mode 2: bf16 out, [B,H,dk,L] head-split + T   (V transposed for PV wmma)
__global__ __launch_bounds__(128) void gemm_xWT(
    const bf16_t* __restrict__ A, const bf16_t* __restrict__ W,
    const float* __restrict__ bias, void* __restrict__ outp,
    int M, int N, int K, int mode) {
  int lane = threadIdx.x & 31;
  int wave = threadIdx.x >> 5;
  int rowBase = blockIdx.x * 128 + (wave >> 1) * 64;
  int colBase = blockIdx.y * 128 + (wave & 1) * 64;

  v8f acc[4][4] = {};
  for (int k0 = 0; k0 < K; k0 += 32) {
    // prefetch rows of the k-tile 8 steps ahead (k0+256) into cache (global_prefetch_b8)
    if (k0 + 256 < K) {
      __builtin_prefetch(A + (size_t)(rowBase + (lane & 15)) * K + k0 + 256, 0, 1);
      __builtin_prefetch(W + (size_t)(colBase + (lane & 15)) * K + k0 + 256, 0, 1);
    }
    v16bf b[4];
#pragma unroll
    for (int j = 0; j < 4; ++j)
      b[j] = load_b(W + (size_t)(colBase + j * 16) * K + k0, K, lane);
#pragma unroll
    for (int i = 0; i < 4; ++i) {
      v16bf a = load_a(A + (size_t)(rowBase + i * 16) * K + k0, K, lane);
#pragma unroll
      for (int j = 0; j < 4; ++j)
        acc[i][j] = wmma_bf16(a, b[j], acc[i][j]);
    }
  }

  // Fully-unrolled epilogue: constant indices into acc => stays in VGPRs.
  int half = lane >> 4, nl = lane & 15;
  if (mode == 0) {
    float* of = (float*)outp;
#pragma unroll
    for (int ms = 0; ms < 4; ++ms)
#pragma unroll
      for (int ns = 0; ns < 4; ++ns) {
        int col = colBase + ns * 16 + nl;
        float bv = bias[col];
#pragma unroll
        for (int r = 0; r < 8; ++r) {
          int row = rowBase + ms * 16 + half * 8 + r;
          of[(size_t)row * N + col] = acc[ms][ns][r] + bv;
        }
      }
  } else if (mode == 1) {
    bf16_t* o = (bf16_t*)outp;
#pragma unroll
    for (int ms = 0; ms < 4; ++ms)
#pragma unroll
      for (int ns = 0; ns < 4; ++ns) {
        int col = colBase + ns * 16 + nl;
        int h = col >> 6, d = col & (DK - 1);
        float bv = bias[col];
#pragma unroll
        for (int r = 0; r < 8; ++r) {
          int row = rowBase + ms * 16 + half * 8 + r;
          int b_ = row >> 11, l = row & (SEQ - 1);
          o[(((size_t)(b_ * N_HEADS + h)) * SEQ + l) * DK + d] =
              (bf16_t)(acc[ms][ns][r] + bv);
        }
      }
  } else {
    bf16_t* o = (bf16_t*)outp;
#pragma unroll
    for (int ms = 0; ms < 4; ++ms)
#pragma unroll
      for (int ns = 0; ns < 4; ++ns) {
        int col = colBase + ns * 16 + nl;
        int h = col >> 6, d = col & (DK - 1);
        float bv = bias[col];
#pragma unroll
        for (int r = 0; r < 8; ++r) {
          int row = rowBase + ms * 16 + half * 8 + r;
          int b_ = row >> 11, l = row & (SEQ - 1);
          o[(((size_t)(b_ * N_HEADS + h)) * DK + d) * SEQ + l] =
              (bf16_t)(acc[ms][ns][r] + bv);
        }
      }
  }
}

// --------------------------- fused flash attention -------------------------
// One wave per 16 query rows; online softmax over 64-key tiles.
// Intervention: key-0 probability halved in the P@V numerator only.
__global__ __launch_bounds__(256) void attn_kernel(
    const bf16_t* __restrict__ Q,   // [B,H,L,dk]
    const bf16_t* __restrict__ Km,  // [B,H,L,dk]
    const bf16_t* __restrict__ Vt,  // [B,H,dk,L]
    bf16_t* __restrict__ O) {       // [B,L,D] bf16 (head-concat)
  __shared__ __align__(16) bf16_t p_lds[8 * 16 * 64];   // 16 KB
  int lane = threadIdx.x & 31;
  int wave = threadIdx.x >> 5;
  int flat = blockIdx.x * 8 + wave;
  int bh = flat >> 7;                 // 128 q-chunks of 16 per (b,h)
  int q0 = (flat & 127) << 4;
  const bf16_t* Qp = Q  + ((size_t)bh * SEQ + q0) * DK;
  const bf16_t* Kp = Km + (size_t)bh * SEQ * DK;
  const bf16_t* Vp = Vt + (size_t)bh * DK * SEQ;
  bf16_t* myP = p_lds + wave * (16 * 64);

  v16bf aq0 = load_a(Qp,      DK, lane);   // Q 16x64 as two 16x32 A-frags
  v16bf aq1 = load_a(Qp + 32, DK, lane);

  int half = lane >> 4, nl = lane & 15;
  const float scale = 0.125f;              // 1/sqrt(64)
  float mrow[8], lrow[8];
  v8f o[4] = {};
#pragma unroll
  for (int r = 0; r < 8; ++r) { mrow[r] = -1e30f; lrow[r] = 0.f; }

  for (int kt = 0; kt < SEQ; kt += 64) {
    // S = Q @ K^T for 64 keys: 4 N-frags x 2 k-steps
    v8f s[4] = {};
#pragma unroll
    for (int f = 0; f < 4; ++f) {
      v16bf b0 = load_b(Kp + (size_t)(kt + f * 16) * DK,      DK, lane);
      s[f] = wmma_bf16(aq0, b0, s[f]);
      v16bf b1 = load_b(Kp + (size_t)(kt + f * 16) * DK + 32, DK, lane);
      s[f] = wmma_bf16(aq1, b1, s[f]);
    }
    // per-row max over this tile (scale > 0 commutes with max)
    float cand[8];
#pragma unroll
    for (int r = 0; r < 8; ++r)
      cand[r] = fmaxf(fmaxf(s[0][r], s[1][r]), fmaxf(s[2][r], s[3][r])) * scale;
#pragma unroll
    for (int off = 1; off < 16; off <<= 1)
#pragma unroll
      for (int r = 0; r < 8; ++r)
        cand[r] = fmaxf(cand[r], __shfl_xor(cand[r], off, 32));

    float mnew[8], corr[8], rsum[8];
#pragma unroll
    for (int r = 0; r < 8; ++r) {
      mnew[r] = fmaxf(mrow[r], cand[r]);
      corr[r] = __expf(mrow[r] - mnew[r]);
      rsum[r] = 0.f;
    }
    // P = exp(S*scale - m); sum uses full P, P@V numerator halves key 0
#pragma unroll
    for (int f = 0; f < 4; ++f)
#pragma unroll
      for (int r = 0; r < 8; ++r) {
        float p = __expf(s[f][r] * scale - mnew[r]);
        rsum[r] += p;
        float pv = (kt == 0 && f == 0 && nl == 0) ? p * 0.5f : p;
        myP[(half * 8 + r) * 64 + f * 16 + nl] = (bf16_t)pv;
      }
#pragma unroll
    for (int off = 1; off < 16; off <<= 1)
#pragma unroll
      for (int r = 0; r < 8; ++r)
        rsum[r] += __shfl_xor(rsum[r], off, 32);
#pragma unroll
    for (int r = 0; r < 8; ++r) {
      lrow[r] = lrow[r] * corr[r] + rsum[r];
      mrow[r] = mnew[r];
    }
#pragma unroll
    for (int f = 0; f < 4; ++f)
#pragma unroll
      for (int r = 0; r < 8; ++r)
        o[f][r] *= corr[r];

    // LDS round-trip converts P from C-layout to A-fragment layout (same wave:
    // DS pipeline is in-order; make the store->load hazard explicit anyway).
    asm volatile("s_wait_dscnt 0x0" ::: "memory");
#pragma unroll
    for (int kk = 0; kk < 64; kk += 32) {
      v16bf ap = load_a(myP + kk, 64, lane);
#pragma unroll
      for (int f = 0; f < 4; ++f) {
        v16bf bv = load_b(Vp + (size_t)(f * 16) * SEQ + kt + kk, SEQ, lane);
        o[f] = wmma_bf16(ap, bv, o[f]);
      }
    }
  }

  int b = bh >> 4, h = bh & 15;
#pragma unroll
  for (int f = 0; f < 4; ++f)
#pragma unroll
    for (int r = 0; r < 8; ++r) {
      int row = q0 + half * 8 + r;
      float v = o[f][r] / lrow[r];
      O[((size_t)(b * SEQ + row)) * D_MODEL + h * DK + f * 16 + nl] = (bf16_t)v;
    }
}

// ---------------------------------------------------------------------------
extern "C" void kernel_launch(void* const* d_in, const int* in_sizes, int n_in,
                              void* d_out, int out_size, void* d_ws, size_t ws_size,
                              hipStream_t stream) {
  const float* x  = (const float*)d_in[0];
  const float* Wq = (const float*)d_in[1];
  const float* bq = (const float*)d_in[2];
  const float* Wk = (const float*)d_in[3];
  const float* bk = (const float*)d_in[4];
  const float* Wv = (const float*)d_in[5];
  const float* bv = (const float*)d_in[6];
  const float* Wo = (const float*)d_in[7];
  const float* bo = (const float*)d_in[8];
  float* out = (float*)d_out;

  char* w = (char*)d_ws;                       // 48 MB used
  bf16_t* xb  = (bf16_t*)(w);                  // x bf16        8 MB
  bf16_t* wqb = (bf16_t*)(w + (8u  << 20));    // weights bf16  2 MB each
  bf16_t* wkb = (bf16_t*)(w + (10u << 20));
  bf16_t* wvb = (bf16_t*)(w + (12u << 20));
  bf16_t* wob = (bf16_t*)(w + (14u << 20));
  bf16_t* qb  = (bf16_t*)(w + (16u << 20));    // Q [B,H,L,dk]  8 MB
  bf16_t* kb  = (bf16_t*)(w + (24u << 20));    // K [B,H,L,dk]  8 MB
  bf16_t* vtb = (bf16_t*)(w + (32u << 20));    // V^T [B,H,dk,L] 8 MB
  bf16_t* ob  = (bf16_t*)(w + (40u << 20));    // attn out bf16 8 MB

  const int nx = ROWS * D_MODEL;      // 4M
  const int nw = D_MODEL * D_MODEL;   // 1M
  f32_to_bf16_kernel<<<(nx + 255) / 256, 256, 0, stream>>>(x,  xb,  nx);
  f32_to_bf16_kernel<<<(nw + 255) / 256, 256, 0, stream>>>(Wq, wqb, nw);
  f32_to_bf16_kernel<<<(nw + 255) / 256, 256, 0, stream>>>(Wk, wkb, nw);
  f32_to_bf16_kernel<<<(nw + 255) / 256, 256, 0, stream>>>(Wv, wvb, nw);
  f32_to_bf16_kernel<<<(nw + 255) / 256, 256, 0, stream>>>(Wo, wob, nw);

  dim3 g(ROWS / 128, D_MODEL / 128);  // 32 x 8 = 256 blocks, 4 waves each
  gemm_xWT<<<g, 128, 0, stream>>>(xb, wqb, bq, qb,  ROWS, D_MODEL, D_MODEL, 1);
  gemm_xWT<<<g, 128, 0, stream>>>(xb, wkb, bk, kb,  ROWS, D_MODEL, D_MODEL, 1);
  gemm_xWT<<<g, 128, 0, stream>>>(xb, wvb, bv, vtb, ROWS, D_MODEL, D_MODEL, 2);

  // 2*16 heads * 128 q-chunks of 16 rows / 8 waves per block = 512 blocks
  attn_kernel<<<(BATCH * N_HEADS * (SEQ / 16)) / 8, 256, 0, stream>>>(qb, kb, vtb, ob);

  gemm_xWT<<<g, 128, 0, stream>>>(ob, wob, bo, out, ROWS, D_MODEL, D_MODEL, 0);
}
